// AttentionMarketEncoder_41343355191539
// MI455X (gfx1250) — matrix operand order repaired and checked by reading, not compiled, not run on hardware
//
#include <hip/hip_runtime.h>

#define WND 180
#define BB  256
#define DM  256
#define NH  8
#define DKH 32
#define NL  4

typedef __attribute__((ext_vector_type(16))) _Float16 v16h;
typedef __attribute__((ext_vector_type(8)))  _Float16 v8h;
typedef __attribute__((ext_vector_type(8)))  float    v8f;

union HV16 { v16h v; v8h h[2]; };

// A-fragment (16x32 f16): lane m=lane&15, K-half=lane>>4.
// elements 0..7  -> K = kh*8 + e         (contiguous 8 halfs)
// elements 8..15 -> K = 16 + kh*8 + e-8  (contiguous 8 halfs)
__device__ __forceinline__ v16h frag_a(const _Float16* p, int row0, int ld, int c0) {
  const int lane = threadIdx.x & 31;
  const int m = lane & 15, kh = lane >> 4;
  const _Float16* q = p + (row0 + m) * ld + c0 + kh * 8;
  HV16 u;
  u.h[0] = *(const v8h*)(q);
  u.h[1] = *(const v8h*)(q + 16);
  return u.v;
}

// B-fragment (32x16 f16) read from a transposed [N][K] buffer:
// lane n=lane&15, K-half=lane>>4; elements 0..7 -> K = kh*8 + e, elements 8..15 -> K = 16 + kh*8 + e-8
__device__ __forceinline__ v16h frag_b(const _Float16* p, int n0, int ld, int k0) {
  const int lane = threadIdx.x & 31;
  const int n = lane & 15, kh = lane >> 4;
  const _Float16* q = p + (n0 + n) * ld + k0 + kh * 8;
  HV16 u;
  u.h[0] = *(const v8h*)(q);
  u.h[1] = *(const v8h*)(q + 16);
  return u.v;
}

__device__ __forceinline__ v8f wmma_f16(v16h a, v16h b, v8f c) {
  return __builtin_amdgcn_wmma_f32_16x16x32_f16(false, a, false, b, (short)0, c, false, false);
}

// reductions across the 16-lane half-groups (wave32): xor masks 1,2,4,8
__device__ __forceinline__ float red_max16(float v) {
  #pragma unroll
  for (int m = 1; m < 16; m <<= 1) v = fmaxf(v, __shfl_xor(v, m, 32));
  return v;
}
__device__ __forceinline__ float red_sum16(float v) {
  #pragma unroll
  for (int m = 1; m < 16; m <<= 1) v += __shfl_xor(v, m, 32);
  return v;
}

// ---- weight prep: f32 -> f16, transposed to [out][in] so B-fragments are contiguous ----
__global__ __launch_bounds__(256) void ame_prep(
    const float* __restrict__ WQ, const float* __restrict__ WK, const float* __restrict__ WV,
    const float* __restrict__ WO, const float* __restrict__ fc1, const float* __restrict__ fc2,
    _Float16* __restrict__ WqT, _Float16* __restrict__ WkT, _Float16* __restrict__ WvT,
    _Float16* __restrict__ WoT, _Float16* __restrict__ fc1T, _Float16* __restrict__ fc2T)
{
  const int idx = blockIdx.x * 256 + threadIdx.x;
  if (idx < NH * DM * DKH) {                       // WQ/WK/WV: [H][D][DK] -> [H][DK][D]
    const int h = idx / (DM * DKH);
    const int rem = idx - h * DM * DKH;
    const int k = rem / DKH;
    const int n = rem - k * DKH;
    const int dst = (h * DKH + n) * DM + k;
    WqT[dst] = (_Float16)WQ[idx];
    WkT[dst] = (_Float16)WK[idx];
    WvT[dst] = (_Float16)WV[idx];
  }
  if (idx < DM * DM) {                             // WO: [D][D] -> [D][D]^T
    const int k = idx / DM, n = idx % DM;
    WoT[n * DM + k] = (_Float16)WO[idx];
  }
  if (idx < NL * DM * DM) {                        // fc1/fc2: [L][D][D] -> transposed
    const int j = idx / (DM * DM);
    const int rem = idx % (DM * DM);
    const int k = rem / DM, n = rem % DM;
    fc1T[j * DM * DM + n * DM + k] = (_Float16)fc1[idx];
    fc2T[j * DM * DM + n * DM + k] = (_Float16)fc2[idx];
  }
}

// ---- main: one workgroup per window slice, 8 waves, all 4 layers in LDS ----
__global__ __launch_bounds__(256, 1) void ame_main(
    const float* __restrict__ mv,
    const float* __restrict__ fcbarW, const float* __restrict__ fcbarB,
    const float* __restrict__ a_gain, const float* __restrict__ a_bias,
    const float* __restrict__ fc1_b,  const float* __restrict__ fc2_b,
    const float* __restrict__ fo_gain, const float* __restrict__ fo_bias,
    const _Float16* __restrict__ WqT, const _Float16* __restrict__ WkT,
    const _Float16* __restrict__ WvT, const _Float16* __restrict__ WoT,
    const _Float16* __restrict__ fc1T, const _Float16* __restrict__ fc2T,
    _Float16* __restrict__ xAll)
{
  extern __shared__ _Float16 smem[];
  _Float16* X  = smem;                  // [256][256] activations (layer input / output)
  _Float16* W1 = X  + DM * BB;          // [256][256] heads concat / 'a' / h1 (in-place)
  _Float16* Qs = W1 + DM * BB;          // [256][32]
  _Float16* Ks = Qs + BB * DKH;         // [256][32]
  _Float16* Vt = Ks + BB * DKH;         // [32][256]  (V transposed: dk-major)
  _Float16* Pb = Vt + DKH * BB;         // [256][32]  softmax P block
  // total = 327680 B = 320 KB (dynamic LDS)

  const int w    = blockIdx.x;
  const int tid  = threadIdx.x;
  const int lane = tid & 31;
  const int wv   = tid >> 5;
  const int mlo  = lane & 15;   // N / column lane
  const int mhi  = lane >> 4;   // row-half of C fragment
  const int r0   = wv * 32;     // this wave owns rows [r0, r0+32)

  const v8f vzero = {0.f,0.f,0.f,0.f,0.f,0.f,0.f,0.f};

  // ---------------- embed: x = [mv, tag] @ fcbar_W + fcbar_b ----------------
  {
    float* fW = (float*)W1;     // stage small weights in LDS
    for (int i = tid; i < 7 * DM; i += 256)
      fW[i] = (i < 6 * DM) ? fcbarW[i] : fcbarB[i - 6 * DM];
    __syncthreads();
    float in6[6];
    const float* mrow = mv + (size_t)(w * BB + tid) * 5;
    #pragma unroll
    for (int f = 0; f < 5; ++f) in6[f] = mrow[f];
    in6[5] = ((float)w - 90.0f) * (1.0f / 90.0f);
    const float* fB = fW + 6 * DM;
    for (int d = 0; d < DM; ++d) {
      float s = fB[d];
      #pragma unroll
      for (int f = 0; f < 6; ++f) s += in6[f] * fW[f * DM + d];
      X[tid * DM + d] = (_Float16)s;
    }
  }
  __syncthreads();

  const float qscale = 0.17677669529663689f;   // 1/sqrt(32), folded into Q

  for (int j = 0; j < NL; ++j) {
    // ======================= multi-head attention =======================
    for (int h = 0; h < NH; ++h) {
      __syncthreads();                      // K/V of previous head no longer in use
      const _Float16* wq = WqT + h * DKH * DM;
      const _Float16* wk = WkT + h * DKH * DM;
      const _Float16* wvp = WvT + h * DKH * DM;
      // --- Q/K/V for own rows: [32 x 256] @ [256 x 32] ---
      #pragma unroll
      for (int mt = 0; mt < 2; ++mt) {
        const int row = r0 + mt * 16;
        v8f aq0 = vzero, aq1 = vzero, ak0 = vzero, ak1 = vzero, av0 = vzero, av1 = vzero;
        for (int kc = 0; kc < 8; ++kc) {
          const v16h aF = frag_a(X, row, DM, kc * 32);
          aq0 = wmma_f16(aF, frag_b(wq,   0, DM, kc * 32), aq0);
          aq1 = wmma_f16(aF, frag_b(wq,  16, DM, kc * 32), aq1);
          ak0 = wmma_f16(aF, frag_b(wk,   0, DM, kc * 32), ak0);
          ak1 = wmma_f16(aF, frag_b(wk,  16, DM, kc * 32), ak1);
          av0 = wmma_f16(aF, frag_b(wvp,  0, DM, kc * 32), av0);
          av1 = wmma_f16(aF, frag_b(wvp, 16, DM, kc * 32), av1);
        }
        #pragma unroll
        for (int r = 0; r < 8; ++r) {
          const int rr = row + mhi * 8 + r;
          Qs[rr * DKH + mlo]        = (_Float16)(aq0[r] * qscale);
          Qs[rr * DKH + 16 + mlo]   = (_Float16)(aq1[r] * qscale);
          Ks[rr * DKH + mlo]        = (_Float16)ak0[r];
          Ks[rr * DKH + 16 + mlo]   = (_Float16)ak1[r];
          Vt[mlo * BB + rr]         = (_Float16)av0[r];   // store V transposed
          Vt[(16 + mlo) * BB + rr]  = (_Float16)av1[r];
        }
      }
      __syncthreads();                      // all rows of K/V visible

      // --- flash attention over 8 key blocks of 32 ---
      float mrun[2][8], lrun[2][8];
      v8f accO[2][2];
      #pragma unroll
      for (int mt = 0; mt < 2; ++mt) {
        accO[mt][0] = vzero; accO[mt][1] = vzero;
        #pragma unroll
        for (int r = 0; r < 8; ++r) { mrun[mt][r] = -1e30f; lrun[mt][r] = 0.f; }
      }
      v16h aQ[2];
      aQ[0] = frag_a(Qs, r0,      DKH, 0);
      aQ[1] = frag_a(Qs, r0 + 16, DKH, 0);
      for (int cb = 0; cb < 8; ++cb) {
        v8f sc[2][2];
        #pragma unroll
        for (int mt = 0; mt < 2; ++mt) {
          sc[mt][0] = wmma_f16(aQ[mt], frag_b(Ks, cb * 32,      DKH, 0), vzero);
          sc[mt][1] = wmma_f16(aQ[mt], frag_b(Ks, cb * 32 + 16, DKH, 0), vzero);
        }
        #pragma unroll
        for (int mt = 0; mt < 2; ++mt) {
          #pragma unroll
          for (int r = 0; r < 8; ++r) {
            float bm = red_max16(fmaxf(sc[mt][0][r], sc[mt][1][r]));
            float mn = fmaxf(mrun[mt][r], bm);
            float corr = __expf(mrun[mt][r] - mn);
            mrun[mt][r] = mn;
            lrun[mt][r] = lrun[mt][r] * corr;
            accO[mt][0][r] = accO[mt][0][r] * corr;
            accO[mt][1][r] = accO[mt][1][r] * corr;
            float p0 = __expf(sc[mt][0][r] - mn);
            float p1 = __expf(sc[mt][1][r] - mn);
            lrun[mt][r] += red_sum16(p0 + p1);
            const int rr = r0 + mt * 16 + mhi * 8 + r;
            Pb[rr * 32 + mlo]      = (_Float16)p0;    // C-layout -> A-layout via LDS
            Pb[rr * 32 + 16 + mlo] = (_Float16)p1;
          }
        }
        #pragma unroll
        for (int mt = 0; mt < 2; ++mt) {
          const v16h aP = frag_a(Pb, r0 + mt * 16, 32, 0);
          accO[mt][0] = wmma_f16(aP, frag_b(Vt,  0, BB, cb * 32), accO[mt][0]);
          accO[mt][1] = wmma_f16(aP, frag_b(Vt, 16, BB, cb * 32), accO[mt][1]);
        }
      }
      // normalize by row sums, write this head's 32 columns of the concat buffer
      #pragma unroll
      for (int mt = 0; mt < 2; ++mt)
        #pragma unroll
        for (int r = 0; r < 8; ++r) {
          const float inv = 1.0f / lrun[mt][r];
          const int rr = r0 + mt * 16 + mhi * 8 + r;
          W1[rr * DM + h * DKH + mlo]      = (_Float16)(accO[mt][0][r] * inv);
          W1[rr * DM + h * DKH + 16 + mlo] = (_Float16)(accO[mt][1][r] * inv);
        }
    } // heads

    // ======== attn out-proj + residual + norm -> 'a' (in place in W1) ========
    #pragma unroll 1
    for (int mt = 0; mt < 2; ++mt) {
      const int row = r0 + mt * 16;
      v8f acc[16];
      #pragma unroll
      for (int nt = 0; nt < 16; ++nt) acc[nt] = vzero;
      for (int kc = 0; kc < 8; ++kc) {
        const v16h aF = frag_a(W1, row, DM, kc * 32);
        #pragma unroll
        for (int nt = 0; nt < 16; ++nt)
          acc[nt] = wmma_f16(aF, frag_b(WoT, nt * 16, DM, kc * 32), acc[nt]);
      }
      float gn[16], bn[16];
      #pragma unroll
      for (int nt = 0; nt < 16; ++nt) {
        gn[nt] = a_gain[j * DM + nt * 16 + mlo] + 1.0f;
        bn[nt] = a_bias[j * DM + nt * 16 + mlo];
      }
      #pragma unroll
      for (int r = 0; r < 8; ++r) {
        const int rr = row + mhi * 8 + r;
        float s1 = 0.f, s2 = 0.f;
        #pragma unroll
        for (int nt = 0; nt < 16; ++nt) {
          float v = acc[nt][r] + (float)X[rr * DM + nt * 16 + mlo];
          acc[nt][r] = v;
          s1 += v; s2 += v * v;
        }
        s1 = red_sum16(s1); s2 = red_sum16(s2);
        const float mean = s1 * (1.0f / 256.0f);
        const float var  = fmaxf((s2 - 256.0f * mean * mean) * (1.0f / 255.0f), 0.f);
        const float inv  = 1.0f / (sqrtf(var) + 1e-9f);
        #pragma unroll
        for (int nt = 0; nt < 16; ++nt)
          W1[rr * DM + nt * 16 + mlo] = (_Float16)((acc[nt][r] - mean) * inv * gn[nt] + bn[nt]);
      }
    }

    // =================== FFN1: h1 = leaky(a @ fc1 + b1) ===================
    const _Float16* f1 = fc1T + j * DM * DM;
    const _Float16* f2 = fc2T + j * DM * DM;
    __builtin_prefetch(f2 + r0 * DM, 0, 1);
    #pragma unroll 1
    for (int mt = 0; mt < 2; ++mt) {
      const int row = r0 + mt * 16;
      v8f acc[16];
      #pragma unroll
      for (int nt = 0; nt < 16; ++nt) acc[nt] = vzero;
      for (int kc = 0; kc < 8; ++kc) {
        const v16h aF = frag_a(W1, row, DM, kc * 32);
        #pragma unroll
        for (int nt = 0; nt < 16; ++nt)
          acc[nt] = wmma_f16(aF, frag_b(f1, nt * 16, DM, kc * 32), acc[nt]);
      }
      float bb[16];
      #pragma unroll
      for (int nt = 0; nt < 16; ++nt) bb[nt] = fc1_b[j * DM + nt * 16 + mlo];
      #pragma unroll
      for (int r = 0; r < 8; ++r) {
        const int rr = row + mhi * 8 + r;
        #pragma unroll
        for (int nt = 0; nt < 16; ++nt) {
          float v = acc[nt][r] + bb[nt];
          v = v > 0.f ? v : 0.01f * v;
          W1[rr * DM + nt * 16 + mlo] = (_Float16)v;
        }
      }
    }

    // ======== FFN2 + residual(x) + norm -> new x (in place in X) ========
    #pragma unroll 1
    for (int mt = 0; mt < 2; ++mt) {
      const int row = r0 + mt * 16;
      v8f acc[16];
      #pragma unroll
      for (int nt = 0; nt < 16; ++nt) acc[nt] = vzero;
      for (int kc = 0; kc < 8; ++kc) {
        const v16h aF = frag_a(W1, row, DM, kc * 32);
        #pragma unroll
        for (int nt = 0; nt < 16; ++nt)
          acc[nt] = wmma_f16(aF, frag_b(f2, nt * 16, DM, kc * 32), acc[nt]);
      }
      float bb[16], gn[16], bn[16];
      #pragma unroll
      for (int nt = 0; nt < 16; ++nt) {
        bb[nt] = fc2_b[j * DM + nt * 16 + mlo];
        gn[nt] = fo_gain[j * DM + nt * 16 + mlo] + 1.0f;
        bn[nt] = fo_bias[j * DM + nt * 16 + mlo];
      }
      #pragma unroll
      for (int r = 0; r < 8; ++r) {
        const int rr = row + mhi * 8 + r;
        float s1 = 0.f, s2 = 0.f;
        #pragma unroll
        for (int nt = 0; nt < 16; ++nt) {
          float v = acc[nt][r] + bb[nt] + (float)X[rr * DM + nt * 16 + mlo];
          acc[nt][r] = v;
          s1 += v; s2 += v * v;
        }
        s1 = red_sum16(s1); s2 = red_sum16(s2);
        const float mean = s1 * (1.0f / 256.0f);
        const float var  = fmaxf((s2 - 256.0f * mean * mean) * (1.0f / 255.0f), 0.f);
        const float inv  = 1.0f / (sqrtf(var) + 1e-9f);
        #pragma unroll
        for (int nt = 0; nt < 16; ++nt)
          X[rr * DM + nt * 16 + mlo] = (_Float16)((acc[nt][r] - mean) * inv * gn[nt] + bn[nt]);
      }
    }
    __syncthreads();
  } // layers

  // ---- dump final activations for this window slice ----
  {
    const uint4* src = (const uint4*)X;
    uint4* dst = (uint4*)(xAll + (size_t)w * BB * DM);
    for (int i = tid; i < BB * DM / 8; i += 256) dst[i] = src[i];
  }
}

// ---- final: out[d][b] = leaky( sum_w x[w][b][d] * fcf[w] + b ) ----
__global__ __launch_bounds__(256) void ame_final(
    const _Float16* __restrict__ xAll, const float* __restrict__ fcfW,
    const float* __restrict__ fcfB, float* __restrict__ out)
{
  __shared__ float fw[WND];
  const int tid = threadIdx.x, b = blockIdx.x;
  if (tid < WND) fw[tid] = fcfW[tid];
  __syncthreads();
  float s = fcfB[0];
  for (int w = 0; w < WND; ++w)
    s += (float)xAll[((size_t)w * BB + b) * DM + tid] * fw[w];
  s = s > 0.f ? s : 0.01f * s;
  out[tid * BB + b] = s;
}

extern "C" void kernel_launch(void* const* d_in, const int* in_sizes, int n_in,
                              void* d_out, int out_size, void* d_ws, size_t ws_size,
                              hipStream_t stream) {
  const float* mv      = (const float*)d_in[0];
  const float* fcbarW  = (const float*)d_in[1];
  const float* fcbarB  = (const float*)d_in[2];
  const float* WQ      = (const float*)d_in[3];
  const float* WK      = (const float*)d_in[4];
  const float* WV      = (const float*)d_in[5];
  const float* WO      = (const float*)d_in[6];
  const float* a_gain  = (const float*)d_in[7];
  const float* a_bias  = (const float*)d_in[8];
  const float* fc1_W   = (const float*)d_in[9];
  const float* fc1_b   = (const float*)d_in[10];
  const float* fc2_W   = (const float*)d_in[11];
  const float* fc2_b   = (const float*)d_in[12];
  const float* fo_gain = (const float*)d_in[13];
  const float* fo_bias = (const float*)d_in[14];
  const float* fcfW    = (const float*)d_in[15];
  const float* fcfB    = (const float*)d_in[16];

  _Float16* ws   = (_Float16*)d_ws;
  _Float16* WqT  = ws;
  _Float16* WkT  = WqT + NH * DKH * DM;
  _Float16* WvT  = WkT + NH * DKH * DM;
  _Float16* WoT  = WvT + NH * DKH * DM;
  _Float16* f1T  = WoT + DM * DM;
  _Float16* f2T  = f1T + NL * DM * DM;
  _Float16* xAll = f2T + NL * DM * DM;   // 180 * 65536 halfs

  ame_prep<<<(NL * DM * DM + 255) / 256, 256, 0, stream>>>(
      WQ, WK, WV, WO, fc1_W, fc2_W, WqT, WkT, WvT, WoT, f1T, f2T);

  const size_t lds = (size_t)(2 * DM * BB + 3 * BB * DKH + DKH * BB) * sizeof(_Float16); // 320 KB
  ame_main<<<WND, 256, lds, stream>>>(
      mv, fcbarW, fcbarB, a_gain, a_bias, fc1_b, fc2_b, fo_gain, fo_bias,
      WqT, WkT, WvT, WoT, f1T, f2T, xAll);

  ame_final<<<BB, 256, 0, stream>>>(xAll, fcfW, fcfB, (float*)d_out);
}
